// PrometheusBlockV2_58488864637020
// MI455X (gfx1250) — compile-verified, matrix operand order checked by reading, not run on hardware
//
#include <hip/hip_runtime.h>
#include <hip/hip_bf16.h>
#include <math.h>

// ---------------- types / constants ----------------
typedef __attribute__((ext_vector_type(16))) __bf16 v16bf;
typedef __attribute__((ext_vector_type(8)))  float  v8f;

struct alignas(16) U4 { unsigned a, b, c, d; };

#define TKN  4096   // B*S tokens
#define DIM  1024
#define HID  2048
#define NEXP 8
#define ROWS 8192   // TKN * top-k(2)

__device__ __forceinline__ __bf16 f2bf(float f) {
  union { float f; unsigned u; } v; v.f = f;
  unsigned r = v.u + 0x7FFFu + ((v.u >> 16) & 1u);
  unsigned short h = (unsigned short)(r >> 16);
  return __builtin_bit_cast(__bf16, h);
}
__device__ __forceinline__ float bf2f(__bf16 b) {
  unsigned short h = __builtin_bit_cast(unsigned short, b);
  union { unsigned u; float f; } v; v.u = ((unsigned)h) << 16;
  return v.f;
}
__device__ __forceinline__ float gelu_exact(float x) {
  return 0.5f * x * (1.0f + erff(x * 0.70710678118654752f));
}

// ---------------- LayerNorm: x -> xn (bf16) ----------------
__global__ __launch_bounds__(256) void ln_kernel(
    const float* __restrict__ x, const float* __restrict__ gamma,
    const float* __restrict__ beta, __bf16* __restrict__ xn) {
  int t = blockIdx.x, tid = threadIdx.x;
  __shared__ float red[256];
  const float* xr = x + (size_t)t * DIM;
  float v[4]; float s = 0.f;
#pragma unroll
  for (int i = 0; i < 4; i++) { v[i] = xr[tid + i * 256]; s += v[i]; }
  red[tid] = s; __syncthreads();
  for (int o = 128; o > 0; o >>= 1) { if (tid < o) red[tid] += red[tid + o]; __syncthreads(); }
  float mu = red[0] * (1.0f / DIM); __syncthreads();
  float s2 = 0.f;
#pragma unroll
  for (int i = 0; i < 4; i++) { float d = v[i] - mu; s2 += d * d; }
  red[tid] = s2; __syncthreads();
  for (int o = 128; o > 0; o >>= 1) { if (tid < o) red[tid] += red[tid + o]; __syncthreads(); }
  float inv = rsqrtf(red[0] * (1.0f / DIM) + 1e-5f);
#pragma unroll
  for (int i = 0; i < 4; i++) {
    int d = tid + i * 256;
    xn[(size_t)t * DIM + d] = f2bf((v[i] - mu) * inv * gamma[d] + beta[d]);
  }
}

// ---------------- transpose + f32->bf16 convert ----------------
__global__ __launch_bounds__(256) void cvt_t_kernel(
    const float* __restrict__ src, __bf16* __restrict__ dst, int R, int C) {
  size_t bofs = (size_t)blockIdx.z * R * C;
  int id = blockIdx.x * 256 + threadIdx.x;
  if (id >= R * C) return;
  int r = id / C, c = id - r * C;
  dst[bofs + (size_t)c * R + r] = f2bf(src[bofs + (size_t)r * C + c]);
}

// ---------------- split f32 -> (hi,lo) bf16 ----------------
__global__ __launch_bounds__(256) void split_kernel(
    const float* __restrict__ s, __bf16* __restrict__ hi,
    __bf16* __restrict__ lo, int n) {
  int id = blockIdx.x * 256 + threadIdx.x; if (id >= n) return;
  float v = s[id];
  __bf16 h = f2bf(v);
  hi[id] = h;
  lo[id] = f2bf(v - bf2f(h));
}
__global__ __launch_bounds__(256) void split_t_kernel(
    const float* __restrict__ s, __bf16* __restrict__ hi,
    __bf16* __restrict__ lo, int N) {
  int id = blockIdx.x * 256 + threadIdx.x; if (id >= N * N) return;
  int r = id / N, c = id - r * N;
  float v = s[(size_t)r * N + c];
  __bf16 h = f2bf(v);
  size_t o = (size_t)c * N + r;
  hi[o] = h;
  lo[o] = f2bf(v - bf2f(h));
}

// ---------------- small elementwise helpers (expm path) ----------------
__global__ void scale_kernel(const float* __restrict__ s, float* __restrict__ d,
                             float a, int n) {
  int id = blockIdx.x * 256 + threadIdx.x; if (id < n) d[id] = s[id] * a;
}
__global__ void axpyI_kernel(const float* __restrict__ s, float* __restrict__ d,
                             float a, int N) {
  int id = blockIdx.x * 256 + threadIdx.x; if (id >= N * N) return;
  int r = id / N, c = id - r * N;
  d[id] = s[id] * a + ((r == c) ? 1.f : 0.f);
}

// ---------------- split-bf16 emulated-f32 WMMA GEMM (expm chain) ----------------
// C = alpha * (A*B) (+ I). A given as hi/lo bf16 row-major [N][K=N],
// B given as hi/lo bf16 transposed [N][K=N]. acc += Ah*Bh + Ah*Bl + Al*Bh.
__global__ __launch_bounds__(256) void wmma_gemm_f32x_kernel(
    const __bf16* __restrict__ Ahi, const __bf16* __restrict__ Alo,
    const __bf16* __restrict__ BThi, const __bf16* __restrict__ BTlo,
    float* __restrict__ C, int N, float alpha, int addI) {
  int m0 = blockIdx.y * 128, n0 = blockIdx.x * 64;
  __shared__ __align__(16) __bf16 sAh[128 * 32];
  __shared__ __align__(16) __bf16 sAl[128 * 32];
  __shared__ __align__(16) __bf16 sBh[64 * 32];
  __shared__ __align__(16) __bf16 sBl[64 * 32];

  int tid = threadIdx.x, lane = tid & 31, wave = tid >> 5;
  int wm = wave & 3, wn = wave >> 2;
  v8f acc[2][2] = {};

  int arow = tid >> 1, ahalf = tid & 1;
  int brow = tid >> 2, bchunk = tid & 3;
  int hl = lane >> 4, rl = lane & 15, klo = hl * 8;

  const __bf16* ArH = Ahi + (size_t)(m0 + arow) * N + ahalf * 16;
  const __bf16* ArL = Alo + (size_t)(m0 + arow) * N + ahalf * 16;
  const __bf16* BrH = BThi + (size_t)(n0 + brow) * N + bchunk * 8;
  const __bf16* BrL = BTlo + (size_t)(n0 + brow) * N + bchunk * 8;

  for (int kk = 0; kk < N; kk += 32) {
    {
      U4* dh = (U4*)(sAh + arow * 32 + ahalf * 16);
      U4* dl = (U4*)(sAl + arow * 32 + ahalf * 16);
      const U4* sh = (const U4*)(ArH + kk);
      const U4* sl = (const U4*)(ArL + kk);
      dh[0] = sh[0]; dh[1] = sh[1];
      dl[0] = sl[0]; dl[1] = sl[1];
      *(U4*)(sBh + brow * 32 + bchunk * 8) = *(const U4*)(BrH + kk);
      *(U4*)(sBl + brow * 32 + bchunk * 8) = *(const U4*)(BrL + kk);
    }
    __syncthreads();

    union Fr { v16bf v; U4 u[2]; } ah[2], al[2], bh[2], bl[2];
#pragma unroll
    for (int i = 0; i < 2; i++) {
      int row = wm * 32 + i * 16 + rl;
      const U4* ph = (const U4*)(sAh + row * 32);
      const U4* pl = (const U4*)(sAl + row * 32);
      ah[i].u[0] = ph[klo >> 3]; ah[i].u[1] = ph[(klo + 16) >> 3];
      al[i].u[0] = pl[klo >> 3]; al[i].u[1] = pl[(klo + 16) >> 3];
    }
#pragma unroll
    for (int j = 0; j < 2; j++) {
      int row = wn * 32 + j * 16 + rl;
      const U4* ph = (const U4*)(sBh + row * 32);
      const U4* pl = (const U4*)(sBl + row * 32);
      bh[j].u[0] = ph[klo >> 3]; bh[j].u[1] = ph[(klo + 16) >> 3];
      bl[j].u[0] = pl[klo >> 3]; bl[j].u[1] = pl[(klo + 16) >> 3];
    }
#pragma unroll
    for (int i = 0; i < 2; i++)
#pragma unroll
      for (int j = 0; j < 2; j++) {
        acc[i][j] = __builtin_amdgcn_wmma_f32_16x16x32_bf16(
            false, ah[i].v, false, bh[j].v, (short)0, acc[i][j], false, false);
        acc[i][j] = __builtin_amdgcn_wmma_f32_16x16x32_bf16(
            false, ah[i].v, false, bl[j].v, (short)0, acc[i][j], false, false);
        acc[i][j] = __builtin_amdgcn_wmma_f32_16x16x32_bf16(
            false, al[i].v, false, bh[j].v, (short)0, acc[i][j], false, false);
      }
    __syncthreads();
  }

#pragma unroll
  for (int i = 0; i < 2; i++)
#pragma unroll
    for (int j = 0; j < 2; j++) {
      int n = n0 + wn * 32 + j * 16 + rl;
#pragma unroll
      for (int r = 0; r < 8; r++) {
        int m = m0 + wm * 32 + i * 16 + hl * 8 + r;
        float v = acc[i][j][r] * alpha + ((addI && m == n) ? 1.f : 0.f);
        C[(size_t)m * N + n] = v;
      }
    }
}

// ---------------- routing: gate logits + top-2 softmax ----------------
__global__ __launch_bounds__(256) void gate_kernel(
    const __bf16* __restrict__ xn, const float* __restrict__ gw,
    const float* __restrict__ gb, int* __restrict__ topi,
    float* __restrict__ topw, int* __restrict__ cnt) {
  int wave = threadIdx.x >> 5, lane = threadIdx.x & 31;
  int t = blockIdx.x * 8 + wave;
  if (t >= TKN) return;
  float acc[NEXP];
#pragma unroll
  for (int e = 0; e < NEXP; e++) acc[e] = 0.f;
  for (int d = lane; d < DIM; d += 32) {
    float xv = bf2f(xn[(size_t)t * DIM + d]);
    const float* g = gw + (size_t)d * NEXP;
#pragma unroll
    for (int e = 0; e < NEXP; e++) acc[e] = fmaf(xv, g[e], acc[e]);
  }
#pragma unroll
  for (int e = 0; e < NEXP; e++)
    for (int o = 16; o > 0; o >>= 1) acc[e] += __shfl_xor(acc[e], o, 32);
  if (lane == 0) {
    float l[NEXP];
#pragma unroll
    for (int e = 0; e < NEXP; e++) l[e] = acc[e] + gb[e];
    int i0 = 0;
#pragma unroll
    for (int e = 1; e < NEXP; e++) if (l[e] > l[i0]) i0 = e;
    int i1 = (i0 == 0) ? 1 : 0;
#pragma unroll
    for (int e = 0; e < NEXP; e++) if (e != i0 && l[e] > l[i1]) i1 = e;
    float p0 = 1.f / (1.f + expf(l[i1] - l[i0]));
    topi[t * 2] = i0; topi[t * 2 + 1] = i1;
    topw[t * 2] = p0; topw[t * 2 + 1] = 1.f - p0;
    atomicAdd(&cnt[i0], 1); atomicAdd(&cnt[i1], 1);
  }
}

__global__ void zero_i32(int* __restrict__ p, int n) {
  int id = blockIdx.x * 256 + threadIdx.x; if (id < n) p[id] = 0;
}

__global__ void offsets_kernel(const int* __restrict__ cnt, int* __restrict__ basev) {
  if (threadIdx.x == 0) {
    int s = 0;
    for (int e = 0; e < NEXP; e++) { basev[e] = s; s += cnt[e]; }
  }
}

__global__ void scatter_kernel(const int* __restrict__ topi, const float* __restrict__ topw,
                               const int* __restrict__ basev, int* __restrict__ pos,
                               int* __restrict__ row_token, float* __restrict__ row_w,
                               int* __restrict__ row_of) {
  int id = blockIdx.x * 256 + threadIdx.x; if (id >= ROWS) return;
  int t = id >> 1, k = id & 1;
  int e = topi[t * 2 + k];
  int slot = atomicAdd(&pos[e], 1);
  int row = basev[e] + slot;
  row_token[row] = t; row_w[row] = topw[t * 2 + k]; row_of[t * 2 + k] = row;
}

// ---------------- bf16 WMMA grouped GEMM (token GEMMs) ----------------
// C[m, n] = act( A[gather(m), :K] * BT[e][n, :K]^T + bias[e][n] ) * row_scale[m]
// Block = 256 threads = 8 wave32; block tile 128x64; wave tile 32x32 (2x2 wmma).
__global__ __launch_bounds__(256) void wmma_gemm_kernel(
    const __bf16* __restrict__ A, int lda,
    const __bf16* __restrict__ BT, unsigned long long bstride,
    const float* __restrict__ bias, int biasStride,
    const int* __restrict__ row_idx, const float* __restrict__ row_scale,
    const int* __restrict__ seg_base, const int* __restrict__ seg_cnt,
    int Mconst, int N, int Kd, void* __restrict__ Cout, int flags) {
  int e = blockIdx.z;
  int base = seg_base ? seg_base[e] : 0;
  int cnt  = seg_cnt  ? seg_cnt[e]  : Mconst;
  int m0 = base + blockIdx.y * 128;
  int mEnd = base + cnt;
  if (m0 >= mEnd) return;
  int n0 = blockIdx.x * 64;
  const __bf16* Bp = BT + (size_t)e * bstride;
  const float* bp = bias ? (bias + (size_t)e * biasStride) : nullptr;

  __shared__ __align__(16) __bf16 sA[128 * 32];
  __shared__ __align__(16) __bf16 sB[64 * 32];

  int tid = threadIdx.x, lane = tid & 31, wave = tid >> 5;
  int wm = wave & 3, wn = wave >> 2;

  v8f acc[2][2] = {};

  int arow = tid >> 1, ahalf = tid & 1;      // A tile: 128 rows, 2 halves of 16 elems
  int brow = tid >> 2, bchunk = tid & 3;     // B tile: 64 rows, 4 chunks of 8 elems
  int hl = lane >> 4, rl = lane & 15, klo = hl * 8;

  // Hoist gather + base pointers out of the K loop (loop-invariant).
  int m_l = m0 + arow; if (m_l > mEnd - 1) m_l = mEnd - 1;
  int ar = row_idx ? row_idx[m_l] : m_l;
  const __bf16* Arow = A + (size_t)ar * lda + ahalf * 16;
  const __bf16* Brow = Bp + (size_t)(n0 + brow) * Kd + bchunk * 8;
  U4* sAdst = (U4*)(sA + arow * 32 + ahalf * 16);
  U4* sBdst = (U4*)(sB + brow * 32 + bchunk * 8);

  for (int kk = 0; kk < Kd; kk += 32) {
    {
      const U4* src = (const U4*)(Arow + kk);
      sAdst[0] = src[0]; sAdst[1] = src[1];
      *sBdst = *(const U4*)(Brow + kk);
    }
    __syncthreads();

    union Fr { v16bf v; U4 u[2]; } a[2], b[2];
#pragma unroll
    for (int i = 0; i < 2; i++) {
      int row = wm * 32 + i * 16 + rl;
      const U4* p = (const U4*)(sA + row * 32);
      a[i].u[0] = p[klo >> 3];
      a[i].u[1] = p[(klo + 16) >> 3];
    }
#pragma unroll
    for (int j = 0; j < 2; j++) {
      int row = wn * 32 + j * 16 + rl;
      const U4* p = (const U4*)(sB + row * 32);
      b[j].u[0] = p[klo >> 3];
      b[j].u[1] = p[(klo + 16) >> 3];
    }
#pragma unroll
    for (int i = 0; i < 2; i++)
#pragma unroll
      for (int j = 0; j < 2; j++)
        acc[i][j] = __builtin_amdgcn_wmma_f32_16x16x32_bf16(
            false, a[i].v, false, b[j].v, (short)0, acc[i][j], false, false);
    __syncthreads();
  }

  // Epilogue: C layout — lane<16: VGPR r <-> M=r, N=lane; lane>=16: M=8+r, N=lane-16
#pragma unroll
  for (int i = 0; i < 2; i++)
#pragma unroll
    for (int j = 0; j < 2; j++) {
      int n = n0 + wn * 32 + j * 16 + rl;
      float bv = bp ? bp[n] : 0.f;
#pragma unroll
      for (int r = 0; r < 8; r++) {
        int m = m0 + wm * 32 + i * 16 + hl * 8 + r;
        if (m < mEnd) {
          float v = acc[i][j][r] + bv;
          if (flags & 1) v = gelu_exact(v);
          if (row_scale) v *= row_scale[m];
          if (flags & 2) ((__bf16*)Cout)[(size_t)m * N + n] = f2bf(v);
          else           ((float*)Cout)[(size_t)m * N + n] = v;
        }
      }
    }
}

// ---------------- combine 2 expert rows per token ----------------
__global__ void combine_kernel(const float* __restrict__ eo, const int* __restrict__ row_of,
                               __bf16* __restrict__ moesum) {
  int id = blockIdx.x * 256 + threadIdx.x;
  int t = id >> 10, d = id & 1023;
  float s = eo[(size_t)row_of[t * 2] * DIM + d] + eo[(size_t)row_of[t * 2 + 1] * DIM + d];
  moesum[(size_t)t * DIM + d] = f2bf(s);
}

// ---------------- gated residual mix ----------------
__global__ __launch_bounds__(256) void mix_kernel(
    const float* __restrict__ x, const float* __restrict__ abel,
    const float* __restrict__ moer, const float* __restrict__ gcw,
    const float* __restrict__ gcb, float* __restrict__ out) {
  int t = blockIdx.x, tid = threadIdx.x;
  __shared__ float red[256];
  float av[4], mv[4], xv[4]; float p = 0.f;
#pragma unroll
  for (int i = 0; i < 4; i++) {
    int d = tid + i * 256;
    av[i] = abel[(size_t)t * DIM + d];
    mv[i] = moer[(size_t)t * DIM + d];
    xv[i] = x[(size_t)t * DIM + d];
    p = fmaf(av[i], gcw[d], p);
    p = fmaf(mv[i], gcw[DIM + d], p);
  }
  red[tid] = p; __syncthreads();
  for (int o = 128; o > 0; o >>= 1) { if (tid < o) red[tid] += red[tid + o]; __syncthreads(); }
  float g = 1.f / (1.f + expf(-(red[0] + gcb[0])));
#pragma unroll
  for (int i = 0; i < 4; i++) {
    int d = tid + i * 256;
    out[(size_t)t * DIM + d] = xv[i] + av[i] * (1.f - g) + mv[i] * g;
  }
}

// ---------------- workspace layout (bytes) ----------------
#define OFF_XN      ((size_t)0)
#define OFF_ABELWT  ((size_t)8388608)
#define OFF_EW1T    ((size_t)10485760)
#define OFF_EW2T    ((size_t)44040192)
#define OFF_RT      ((size_t)77594624)
#define OFF_ABELOUT ((size_t)79691776)
#define OFF_H       ((size_t)96468992)    // aliased as 8x4MiB expm f32 scratch
#define OFF_EO      ((size_t)130023424)   // aliased as 4x2MiB split-bf16 scratch
#define OFF_MOESUM  ((size_t)163577856)
#define OFF_MOEROT  ((size_t)171966464)
#define OFF_ROUTE   ((size_t)188743680)

extern "C" void kernel_launch(void* const* d_in, const int* in_sizes, int n_in,
                              void* d_out, int out_size, void* d_ws, size_t ws_size,
                              hipStream_t stream) {
  (void)in_sizes; (void)n_in; (void)out_size; (void)ws_size;
  const float* x        = (const float*)d_in[0];
  const float* ln_gamma = (const float*)d_in[1];
  const float* ln_beta  = (const float*)d_in[2];
  const float* abel_w   = (const float*)d_in[3];
  const float* abel_b   = (const float*)d_in[4];
  const float* gate_w   = (const float*)d_in[5];
  const float* gate_b   = (const float*)d_in[6];
  const float* ew1      = (const float*)d_in[7];
  const float* eb1      = (const float*)d_in[8];
  const float* ew2      = (const float*)d_in[9];
  const float* eb2      = (const float*)d_in[10];
  const float* gens     = (const float*)d_in[11];
  const float* gc_w     = (const float*)d_in[12];
  const float* gc_b     = (const float*)d_in[13];
  float* out = (float*)d_out;

  char* ws = (char*)d_ws;
  __bf16* xn      = (__bf16*)(ws + OFF_XN);
  __bf16* abelwT  = (__bf16*)(ws + OFF_ABELWT);
  __bf16* ew1T    = (__bf16*)(ws + OFF_EW1T);
  __bf16* ew2T    = (__bf16*)(ws + OFF_EW2T);
  __bf16* RT      = (__bf16*)(ws + OFF_RT);
  float*  abelout = (float*)(ws + OFF_ABELOUT);
  __bf16* hbuf    = (__bf16*)(ws + OFF_H);
  float*  eo      = (float*)(ws + OFF_EO);
  __bf16* moesum  = (__bf16*)(ws + OFF_MOESUM);
  float*  moerot  = (float*)(ws + OFF_MOEROT);
  int*    ri      = (int*)(ws + OFF_ROUTE);

  int*   cntp   = ri;
  int*   basev  = ri + 8;
  int*   pos    = ri + 16;
  int*   topi   = ri + 24;
  float* topw   = (float*)(ri + 24 + 8192);
  int*   row_token = ri + 24 + 16384;
  float* row_w  = (float*)(ri + 24 + 24576);
  int*   row_of = ri + 24 + 32768;

  // expm f32 scratch aliased over hbuf (hbuf written later)
  float* exb[8];
  for (int i = 0; i < 8; i++) exb[i] = (float*)(ws + OFF_H + (size_t)i * 4194304);
  // split-bf16 scratch aliased over eo (eo written later)
  __bf16* spAhi = (__bf16*)(ws + OFF_EO);
  __bf16* spAlo = (__bf16*)(ws + OFF_EO + 2097152);
  __bf16* spBhi = (__bf16*)(ws + OFF_EO + 4194304);
  __bf16* spBlo = (__bf16*)(ws + OFF_EO + 6291456);

  // 1) zero routing counters
  zero_i32<<<1, 32, 0, stream>>>(ri, 24);

  // 2) LayerNorm -> xn bf16
  ln_kernel<<<TKN, 256, 0, stream>>>(x, ln_gamma, ln_beta, xn);

  // 3) weight convert+transpose to bf16 [N][K]
  cvt_t_kernel<<<dim3((DIM * DIM) / 256, 1, 1), 256, 0, stream>>>(abel_w, abelwT, DIM, DIM);
  cvt_t_kernel<<<dim3((DIM * HID) / 256, 1, NEXP), 256, 0, stream>>>(ew1, ew1T, DIM, HID);
  cvt_t_kernel<<<dim3((HID * DIM) / 256, 1, NEXP), 256, 0, stream>>>(ew2, ew2T, HID, DIM);

  // 4) R = expm(g0)*expm(g1)*expm(g2): scaling(2^-6) + Horner(8) + 6 squarings.
  //    All 1024^3 GEMMs run on WMMA units via split-bf16 emulated f32.
  {
    auto gemm_f32x = [&](const float* A, const float* Bm, float* Cm,
                         float alpha, int addI) {
      split_kernel<<<(DIM * DIM) / 256, 256, 0, stream>>>(A, spAhi, spAlo, DIM * DIM);
      split_t_kernel<<<(DIM * DIM) / 256, 256, 0, stream>>>(Bm, spBhi, spBlo, DIM);
      wmma_gemm_f32x_kernel<<<dim3(DIM / 64, DIM / 128), 256, 0, stream>>>(
          spAhi, spAlo, spBhi, spBlo, Cm, DIM, alpha, addI);
    };
    float* Tm = exb[0];
    float* P  = exb[1];
    float* Q  = exb[2];
    const int q = 8, ssq = 6;
    for (int r = 0; r < 3; r++) {
      const float* G = gens + (size_t)r * DIM * DIM;
      scale_kernel<<<(DIM * DIM) / 256, 256, 0, stream>>>(G, Tm, 1.0f / 64.0f, DIM * DIM);
      axpyI_kernel<<<(DIM * DIM) / 256, 256, 0, stream>>>(Tm, P, 1.0f / (float)q, DIM);
      for (int k = q - 1; k >= 1; k--) {
        gemm_f32x(Tm, P, Q, 1.0f / (float)k, 1);
        float* t2 = P; P = Q; Q = t2;
      }
      float* Er = exb[3 + r];
      for (int j = 0; j < ssq; j++) {
        float* dst = (j == ssq - 1) ? Er : Q;
        gemm_f32x(P, P, dst, 1.0f, 0);
        if (j < ssq - 1) { float* t2 = P; P = Q; Q = t2; }
      }
    }
    gemm_f32x(exb[3], exb[4], exb[6], 1.0f, 0);
    gemm_f32x(exb[6], exb[5], exb[7], 1.0f, 0);
    cvt_t_kernel<<<dim3((DIM * DIM) / 256, 1, 1), 256, 0, stream>>>(exb[7], RT, DIM, DIM);
  }

  // 5) gate + top-2 routing
  gate_kernel<<<TKN / 8, 256, 0, stream>>>(xn, gate_w, gate_b, topi, topw, cntp);
  offsets_kernel<<<1, 32, 0, stream>>>(cntp, basev);
  scatter_kernel<<<ROWS / 256, 256, 0, stream>>>(topi, topw, basev, pos,
                                                 row_token, row_w, row_of);

  // 6) MoE up-projection: h = gelu(xn[tok] @ ew1[e] + eb1[e])   (bf16 out)
  wmma_gemm_kernel<<<dim3(HID / 64, TKN / 128, NEXP), 256, 0, stream>>>(
      xn, DIM, ew1T, (unsigned long long)HID * DIM, eb1, HID,
      row_token, nullptr, basev, cntp, 0, HID, DIM, (void*)hbuf, /*gelu|bf16*/ 3);

  // 7) MoE down-projection, scaled by routing weight: eo = (h @ ew2[e] + eb2[e]) * w
  wmma_gemm_kernel<<<dim3(DIM / 64, TKN / 128, NEXP), 256, 0, stream>>>(
      hbuf, HID, ew2T, (unsigned long long)DIM * HID, eb2, DIM,
      nullptr, row_w, basev, cntp, 0, DIM, HID, (void*)eo, 0);

  // 8) sum the two expert rows per token -> moesum bf16
  combine_kernel<<<(TKN * DIM) / 256, 256, 0, stream>>>(eo, row_of, moesum);

  // 9) abel branch: gelu(xn @ abel_w + abel_b) -> f32
  wmma_gemm_kernel<<<dim3(DIM / 64, TKN / 128, 1), 256, 0, stream>>>(
      xn, DIM, abelwT, 0ULL, abel_b, 0,
      nullptr, nullptr, nullptr, nullptr, TKN, DIM, DIM, (void*)abelout, 1);

  // 10) apply fused rotation: moerot = moesum @ R
  wmma_gemm_kernel<<<dim3(DIM / 64, TKN / 128, 1), 256, 0, stream>>>(
      moesum, DIM, RT, 0ULL, nullptr, 0,
      nullptr, nullptr, nullptr, nullptr, TKN, DIM, DIM, (void*)moerot, 0);

  // 11) sigmoid-gated mix + residual -> d_out
  mix_kernel<<<TKN, 256, 0, stream>>>(x, abelout, moerot, gc_w, gc_b, out);
}
